// MultiHeadAttention_14164802142523
// MI455X (gfx1250) — matrix operand order repaired and checked by reading, not compiled, not run on hardware
//
#include <hip/hip_runtime.h>
#include <math.h>
#include <stdint.h>

// ---------------- problem constants ----------------
#define Bc   2
#define Sc   2048
#define Hc   2048
#define NHc  16
#define HDc  128
#define Mrows (Bc * Sc)       // 4096

typedef __attribute__((ext_vector_type(16))) __bf16 v16bf;
typedef __attribute__((ext_vector_type(8)))  float  v8f;

// ---------------- WMMA helpers ----------------
__device__ __forceinline__ v8f wmma_bf16(v16bf a, v16bf b, v8f c) {
  return __builtin_amdgcn_wmma_f32_16x16x32_bf16(false, a, false, b, (short)0, c,
                                                 false, false);
}

// A fragment: 16x32 bf16, element (m,k) at p[m*ld + k]
__device__ __forceinline__ v16bf load_a16x32(const __bf16* p, int ld, int lane) {
  const int row = lane & 15;
  const int kb  = (lane >> 4) << 4;  // 0 or 16
  const __bf16* r = p + row * ld + kb;
  v16bf f;
#pragma unroll
  for (int i = 0; i < 8; ++i) {
    f[i]     = r[i];
    f[8 + i] = r[8 + i];
  }
  return f;
}

// B fragment: 32x16 bf16, element (k,n) at p[k*ld + n]
__device__ __forceinline__ v16bf load_b32x16(const __bf16* p, int ld, int lane) {
  const int n  = lane & 15;
  const int kb = (lane >> 4) << 4;  // 0 or 16
  v16bf f;
#pragma unroll
  for (int i = 0; i < 16; ++i) f[i] = p[(kb + i) * ld + n];
  return f;
}

// B fragment from transposed source: B[k,n] = src[n*ld + k]  (for Q·K^T)
__device__ __forceinline__ v16bf load_b32x16_T(const __bf16* p, int ld, int lane) {
  const int n  = lane & 15;
  const int kb = (lane >> 4) << 4;
  const __bf16* r = p + n * ld + kb;
  v16bf f;
#pragma unroll
  for (int i = 0; i < 16; ++i) f[i] = r[i];
  return f;
}

// ---------------- CDNA5 async global -> LDS copy ----------------
// ldsoff: byte offset into LDS (low 32 bits of generic shared address)
__device__ __forceinline__ void async_ld_b128(uint32_t ldsoff, const void* gaddr) {
  asm volatile("global_load_async_to_lds_b128 %0, %1, off"
               :: "v"(ldsoff), "v"((uint64_t)(uintptr_t)gaddr)
               : "memory");
}
__device__ __forceinline__ void wait_async0() {
  asm volatile("s_wait_asynccnt 0x0" ::: "memory");
}
__device__ __forceinline__ uint32_t lds_off(const void* p) {
  return (uint32_t)(uintptr_t)p;  // LDS aperture: low 32 bits = byte offset
}

// ---------------- fp32 -> bf16 conversion ----------------
__global__ __launch_bounds__(256) void cvt_f32_bf16(const float* __restrict__ src,
                                                    __bf16* __restrict__ dst, int n) {
  int i = (blockIdx.x * 256 + threadIdx.x) * 4;
  if (i + 3 < n) {
    float4 v = *(const float4*)(src + i);
    dst[i + 0] = (__bf16)v.x;
    dst[i + 1] = (__bf16)v.y;
    dst[i + 2] = (__bf16)v.z;
    dst[i + 3] = (__bf16)v.w;
  }
}

// ---------------- bf16 WMMA GEMM (async double-buffered) ----------------
// C[M,N] = A[M,K] @ W[K,N] + bias[N]
// MODE 0: write bf16 into [B, NH, S, HD] layout (QKV projections)
// MODE 1: write fp32 row-major (output projection)
#define TM 128
#define TN 64
#define TK 32
#define BUFE (TM * TK + TK * TN)  // 6144 bf16 per buffer

template <int MODE>
__global__ __launch_bounds__(256) void gemm_bf16_kernel(
    const __bf16* __restrict__ A, const __bf16* __restrict__ W,
    const float* __restrict__ bias, void* __restrict__ outp,
    int M, int N, int K) {
  __shared__ __align__(16) __bf16 smem[2 * BUFE];  // 24 KB

  const int tid  = threadIdx.x;
  const int lane = tid & 31;
  const int w    = tid >> 5;
  const int bm   = blockIdx.y * TM;
  const int bn   = blockIdx.x * TN;
  const int wm   = w * 16;  // wave row offset within tile (8 waves x 16 = 128)

  // loader maps (b128 = 8 bf16 per op)
  const int ar0 = tid >> 2, ar1 = (tid + 256) >> 2, ac = (tid & 3) * 8;
  const int br = tid >> 3, bcol = (tid & 7) * 8;

  auto issue = [&](int k0, int buf) {
    const uint32_t la = lds_off(smem + buf * BUFE);
    const uint32_t lb = la + TM * TK * 2;
    async_ld_b128(la + tid * 16, &A[(size_t)(bm + ar0) * K + k0 + ac]);
    async_ld_b128(la + (tid + 256) * 16, &A[(size_t)(bm + ar1) * K + k0 + ac]);
    async_ld_b128(lb + tid * 16, &W[(size_t)(k0 + br) * N + bn + bcol]);
  };

  v8f acc[4];
#pragma unroll
  for (int j = 0; j < 4; ++j) acc[j] = v8f{};

  issue(0, 0);
  const int NT = K / TK;
  for (int t = 0; t < NT; ++t) {
    const int cur = t & 1;
    wait_async0();        // my async writes for tile t have landed in LDS
    __syncthreads();      // everyone arrived; previous compute on other buf done
    if (t + 1 < NT) issue((t + 1) * TK, 1 - cur);

    const __bf16* sA = smem + cur * BUFE;
    const __bf16* sB = sA + TM * TK;
    v16bf a = load_a16x32(&sA[wm * TK], TK, lane);
#pragma unroll
    for (int j = 0; j < 4; ++j) {
      v16bf b = load_b32x16(&sB[j * 16], TN, lane);
      acc[j] = wmma_bf16(a, b, acc[j]);
    }
  }

  const int hi = lane >> 4, nl = lane & 15;
#pragma unroll
  for (int j = 0; j < 4; ++j) {
#pragma unroll
    for (int v = 0; v < 8; ++v) {
      const int row = bm + wm + 8 * hi + v;
      const int col = bn + j * 16 + nl;
      const float r = acc[j][v] + bias[col];
      if (MODE == 0) {
        // remap [b*S + s, nh*HD + hd] -> [b, nh, s, hd]
        __bf16* out = (__bf16*)outp;
        const int b = row >> 11, s = row & (Sc - 1);
        const int nh = col >> 7, hd = col & (HDc - 1);
        out[(((size_t)b * NHc + nh) * Sc + s) * HDc + hd] = (__bf16)r;
      } else {
        ((float*)outp)[(size_t)row * N + col] = r;
      }
    }
  }
}

// ---------------- flash attention (async double-buffered K/V) ----------------
// grid: (S/128, B*NH), block 256 (8 waves). Wave w owns Q rows [16w,16w+16).
__global__ __launch_bounds__(256) void flash_attn_kernel(
    const __bf16* __restrict__ Q, const __bf16* __restrict__ Kh,
    const __bf16* __restrict__ V, const float* __restrict__ mask,
    __bf16* __restrict__ ctx) {
  // 32KB region: first stages the 128x128 Q tile, then is reused as the
  // double-buffered K/V tiles (each buffer: K 32x128 + V 32x128 = 16KB).
  // +8KB per-wave P scratch.
  __shared__ __align__(16) __bf16 smem[16384 + 8 * 16 * 32];  // 40 KB

  const int tid = threadIdx.x, lane = tid & 31, w = tid >> 5;
  const int qt = blockIdx.x;       // 128-row q tile
  const int bh = blockIdx.y;       // b*NH + h
  const int b  = bh >> 4;
  const int h  = bh & (NHc - 1);
  const size_t base = (size_t)bh * Sc * HDc;

  // stage Q tile (128x128 bf16 = 2048 b128 vectors) via async copies
  {
    const __bf16* gq = Q + base + (size_t)(qt * 128) * HDc;
    const uint32_t lq = lds_off(smem);
#pragma unroll
    for (int i = 0; i < 8; ++i) {
      const int vec = tid + i * 256;
      async_ld_b128(lq + vec * 16, gq + vec * 8);
    }
  }
  wait_async0();
  __syncthreads();

  v16bf qf[4];
#pragma unroll
  for (int kk = 0; kk < 4; ++kk)
    qf[kk] = load_a16x32(&smem[(w * 16) * HDc + kk * 32], HDc, lane);
  __syncthreads();  // Q fragments register-resident; smem free for K/V

  auto issue_tile = [&](int kt, int buf) {
    const __bf16* gk = Kh + base + (size_t)(kt * 32) * HDc;
    const __bf16* gv = V + base + (size_t)(kt * 32) * HDc;
    const uint32_t lk = lds_off(smem + buf * 8192);
    const uint32_t lv = lk + 4096 * 2;
    async_ld_b128(lk + tid * 16, gk + tid * 8);
    async_ld_b128(lk + (tid + 256) * 16, gk + (tid + 256) * 8);
    async_ld_b128(lv + tid * 16, gv + tid * 8);
    async_ld_b128(lv + (tid + 256) * 16, gv + (tid + 256) * 8);
  };

  v8f o[8];
#pragma unroll
  for (int j = 0; j < 8; ++j) o[j] = v8f{};
  float m[8], l[8];
#pragma unroll
  for (int v = 0; v < 8; ++v) { m[v] = -1e30f; l[v] = 0.f; }

  const float scale = 0.08838834764831845f;  // 1/sqrt(128)
  const int hi = lane >> 4, nl = lane & 15;
  __bf16* sPw = smem + 16384 + w * (16 * 32);

  issue_tile(0, 0);
  const int NT = Sc / 32;
  for (int kt = 0; kt < NT; ++kt) {
    const int cur = kt & 1;
    wait_async0();        // tile kt landed in LDS (my portion)
    __syncthreads();      // all portions landed; prev compute on other buf done
    if (kt + 1 < NT) issue_tile(kt + 1, 1 - cur);

    const __bf16* sK = smem + cur * 8192;
    const __bf16* sV = sK + 4096;

    // scores: 16x32 = Q(16x128) @ K^T(128x32)
    v8f s0 = {}, s1 = {};
#pragma unroll
    for (int kk = 0; kk < 4; ++kk) {
      v16bf bk0 = load_b32x16_T(&sK[kk * 32], HDc, lane);             // keys 0..15
      v16bf bk1 = load_b32x16_T(&sK[16 * HDc + kk * 32], HDc, lane);  // keys 16..31
      s0 = wmma_bf16(qf[kk], bk0, s0);
      s1 = wmma_bf16(qf[kk], bk1, s1);
    }

    const int kcol = kt * 32;
    const int qrow0 = qt * 128 + w * 16 + 8 * hi;
    float mx[8];
#pragma unroll
    for (int v = 0; v < 8; ++v) {
      const size_t mrow = ((size_t)b * Sc + qrow0 + v) * Sc + kcol;
      float a0 = s0[v] * scale + mask[mrow + nl];
      float a1 = s1[v] * scale + mask[mrow + 16 + nl];
      s0[v] = a0; s1[v] = a1;
      float t = fmaxf(a0, a1);
#pragma unroll
      for (int off = 1; off < 16; off <<= 1) t = fmaxf(t, __shfl_xor(t, off, 32));
      mx[v] = t;
    }

    // online softmax update; stash P (bf16) into wave-private LDS
#pragma unroll
    for (int v = 0; v < 8; ++v) {
      const float mn = fmaxf(m[v], mx[v]);
      const float al = __expf(m[v] - mn);
      const float p0 = __expf(s0[v] - mn);
      const float p1 = __expf(s1[v] - mn);
      float rs = p0 + p1;
#pragma unroll
      for (int off = 1; off < 16; off <<= 1) rs += __shfl_xor(rs, off, 32);
      l[v] = l[v] * al + rs;
      m[v] = mn;
#pragma unroll
      for (int j = 0; j < 8; ++j) o[j][v] *= al;
      sPw[(8 * hi + v) * 32 + nl]      = (__bf16)p0;
      sPw[(8 * hi + v) * 32 + 16 + nl] = (__bf16)p1;
    }

    // wave-private LDS scratch: our own DS stores only, no barrier needed
    asm volatile("s_wait_dscnt 0x0" ::: "memory");

    v16bf pf = load_a16x32(sPw, 32, lane);  // P: 16x32
#pragma unroll
    for (int j = 0; j < 8; ++j) {
      v16bf bv = load_b32x16(&sV[j * 16], HDc, lane);  // V: 32 keys x 16 hd
      o[j] = wmma_bf16(pf, bv, o[j]);
    }
  }

  // normalize and write context back to [b, s, h] bf16
#pragma unroll
  for (int v = 0; v < 8; ++v) {
    const float inv = 1.f / l[v];
    const int srow = qt * 128 + w * 16 + 8 * hi + v;
    const size_t orow = ((size_t)b * Sc + srow) * Hc + (size_t)h * HDc;
#pragma unroll
    for (int j = 0; j < 8; ++j)
      ctx[orow + j * 16 + nl] = (__bf16)(o[j][v] * inv);
  }
}

// ---------------- launcher ----------------
extern "C" void kernel_launch(void* const* d_in, const int* in_sizes, int n_in,
                              void* d_out, int out_size, void* d_ws, size_t ws_size,
                              hipStream_t stream) {
  const float* X   = (const float*)d_in[0];
  const float* msk = (const float*)d_in[1];
  const float* WQ  = (const float*)d_in[2];
  const float* bQ  = (const float*)d_in[3];
  const float* WK  = (const float*)d_in[4];
  const float* bK  = (const float*)d_in[5];
  const float* WV  = (const float*)d_in[6];
  const float* bV  = (const float*)d_in[7];
  const float* WO  = (const float*)d_in[8];
  const float* bO  = (const float*)d_in[9];

  const size_t XE = (size_t)Mrows * Hc;  // 8,388,608
  const size_t WE = (size_t)Hc * Hc;     // 4,194,304

  __bf16* Xb  = (__bf16*)d_ws;
  __bf16* Wqb = Xb + XE;
  __bf16* Wkb = Wqb + WE;
  __bf16* Wvb = Wkb + WE;
  __bf16* Wob = Wvb + WE;
  __bf16* Qb  = Wob + WE;
  __bf16* Kb  = Qb + XE;
  __bf16* Vb  = Kb + XE;
  __bf16* Ctx = Vb + XE;

  // fp32 -> bf16 copies
  cvt_f32_bf16<<<(int)(XE / 1024), 256, 0, stream>>>(X, Xb, (int)XE);
  cvt_f32_bf16<<<(int)(WE / 1024), 256, 0, stream>>>(WQ, Wqb, (int)WE);
  cvt_f32_bf16<<<(int)(WE / 1024), 256, 0, stream>>>(WK, Wkb, (int)WE);
  cvt_f32_bf16<<<(int)(WE / 1024), 256, 0, stream>>>(WV, Wvb, (int)WE);
  cvt_f32_bf16<<<(int)(WE / 1024), 256, 0, stream>>>(WO, Wob, (int)WE);

  // QKV projections -> [B, NH, S, HD] bf16
  dim3 ggrid(Hc / TN, Mrows / TM);
  gemm_bf16_kernel<0><<<ggrid, 256, 0, stream>>>(Xb, Wqb, bQ, (void*)Qb,
                                                 Mrows, Hc, Hc);
  gemm_bf16_kernel<0><<<ggrid, 256, 0, stream>>>(Xb, Wkb, bK, (void*)Kb,
                                                 Mrows, Hc, Hc);
  gemm_bf16_kernel<0><<<ggrid, 256, 0, stream>>>(Xb, Wvb, bV, (void*)Vb,
                                                 Mrows, Hc, Hc);

  // flash attention -> Ctx [b, s, h] bf16
  dim3 fgrid(Sc / 128, Bc * NHc);
  flash_attn_kernel<<<fgrid, 256, 0, stream>>>(Qb, Kb, Vb, msk, Ctx);

  // output projection -> fp32 d_out
  gemm_bf16_kernel<1><<<ggrid, 256, 0, stream>>>(Ctx, Wob, bO, d_out,
                                                 Mrows, Hc, Hc);
}
